// NvFP4LinearLayer_21938692948405
// MI455X (gfx1250) — matrix operand order repaired and checked by reading, not compiled, not run on hardware
//
#include <hip/hip_runtime.h>
#include <hip/hip_bf16.h>
#include <stdint.h>

typedef __attribute__((ext_vector_type(16))) _Float16 v16h;
typedef __attribute__((ext_vector_type(8)))  _Float16 v8h;
typedef __attribute__((ext_vector_type(8)))  float    v8f;

#define K_DIM 4096
#define N_DIM 4096
#define BM    128
#define BN_G  256      // GEMM block tile N
#define BK    32
#define DEPTH 4        // async pipeline depth

// ---------------------------------------------------------------------------
// E2M1 decode helpers
// ---------------------------------------------------------------------------
__device__ __forceinline__ uint32_t e2m1_f16_bits(int code) {
    const uint32_t m = (uint32_t)(code & 7);
    uint32_t bits = (m < 2) ? m * 0x3800u
                            : ((((m >> 1) + 14u) << 10) | ((m & 1u) << 9));
    bits |= ((uint32_t)code & 8u) << 12;
    return bits;
}

__device__ __forceinline__ float e2m1_f32(int code) {
    const uint32_t m = (uint32_t)(code & 7);
    uint32_t bits = (m < 2) ? m * 0x3F000000u
                            : ((((m >> 1) + 126u) << 23) | ((m & 1u) << 22));
    bits |= ((uint32_t)code & 8u) << 28;
    float f;
    __builtin_memcpy(&f, &bits, 4);
    return f;
}

// ---------------------------------------------------------------------------
// Prep kernel 1: x f32 -> f16
// ---------------------------------------------------------------------------
__global__ __launch_bounds__(256)
void cvt_x_f16(const float* __restrict__ x, _Float16* __restrict__ xh) {
    const size_t i = ((size_t)blockIdx.x * 256 + threadIdx.x) * 8;
    const float4 f0 = *(const float4*)(x + i);
    const float4 f1 = *(const float4*)(x + i + 4);
    v8h h;
    h[0] = (_Float16)f0.x; h[1] = (_Float16)f0.y;
    h[2] = (_Float16)f0.z; h[3] = (_Float16)f0.w;
    h[4] = (_Float16)f1.x; h[5] = (_Float16)f1.y;
    h[6] = (_Float16)f1.z; h[7] = (_Float16)f1.w;
    *(v8h*)(xh + i) = h;
}

// ---------------------------------------------------------------------------
// Prep kernel 2: one 16-element NVFP4 block / thread -> f16
// ---------------------------------------------------------------------------
__global__ __launch_bounds__(256)
void dequant_w_f16(const int* __restrict__ wp, const float* __restrict__ ws,
                   _Float16* __restrict__ wh) {
    const size_t blk = (size_t)blockIdx.x * 256 + threadIdx.x;
    const int4* q = (const int4*)(wp + blk * 8);
    const int4 w0 = q[0], w1 = q[1];
    const float scale = ws[blk];
    const int wb[8] = {w0.x, w0.y, w0.z, w0.w, w1.x, w1.y, w1.z, w1.w};
    v8h lo, hi;
#pragma unroll
    for (int e = 0; e < 16; ++e) {
        const int byte = wb[e >> 1];
        const int code = (e & 1) ? (byte & 15) : ((byte >> 4) & 15);
        const _Float16 hv = (_Float16)(e2m1_f32(code) * scale);
        if (e < 8) lo[e] = hv; else hi[e - 8] = hv;
    }
    *(v8h*)(wh + blk * 16)     = lo;
    *(v8h*)(wh + blk * 16 + 8) = hi;
}

// ---------------------------------------------------------------------------
// Async tile copy: 6 ops / thread / tile (A: 2 chunks, B: 4)
// A LDS rows stored in natural K order [0-31] so every WMMA
// fragment is a single contiguous 32B LDS read.
// ---------------------------------------------------------------------------
__device__ __forceinline__ void issue_tile(uint32_t laA, const _Float16* ga,
                                           uint32_t lbB, const _Float16* gb) {
    asm volatile(
        "global_load_async_to_lds_b128 %0, %2, off\n\t"
        "global_load_async_to_lds_b128 %1, %2, off offset:16\n\t"
        "global_load_async_to_lds_b128 %3, %4, off\n\t"
        "global_load_async_to_lds_b128 %3, %4, off offset:16\n\t"
        "global_load_async_to_lds_b128 %3, %4, off offset:32\n\t"
        "global_load_async_to_lds_b128 %3, %4, off offset:48"
        :: "v"(laA), "v"(laA + 16u), "v"(ga), "v"(lbB), "v"(gb)
        : "memory");
}

// ---------------------------------------------------------------------------
// Main GEMM: 128x256 block tile, 64x64 wave tile, depth-4 async pipeline
// ---------------------------------------------------------------------------
__global__ __launch_bounds__(256)
void gemm_f16_wmma(const _Float16* __restrict__ xh,
                   const _Float16* __restrict__ wh,
                   const float* __restrict__ bias,
                   float* __restrict__ out) {
    __shared__ __align__(32) _Float16 sA[DEPTH][BM * BK];   // 4 x 8 KB
    __shared__ __align__(32) _Float16 sB[DEPTH][BN_G * BK]; // 4 x 16 KB

    const int tid   = threadIdx.x;
    const int lane  = tid & 31;
    const int wave  = tid >> 5;
    const int waveM = wave & 1;   // 2 waves along M (64 each)
    const int waveN = wave >> 1;  // 4 waves along N (64 each)
    const int half  = lane >> 4;
    const int fr    = lane & 15;
    const int bM    = blockIdx.y * BM;
    const int bN    = blockIdx.x * BN_G;

    // Staging roles: A -> (row 0..127, K half), B -> full row per thread
    const int aRow = tid >> 1;
    const int kh   = tid & 1;

    const _Float16* aSrc = xh + (size_t)(bM + aRow) * K_DIM + kh * 16;
    const _Float16* bSrc = wh + (size_t)(bN + tid) * K_DIM;

    // Generic->LDS byte offsets (LDS offset lives in addr[31:0]).
    const uint32_t laBase = (uint32_t)(uintptr_t)&sA[0][aRow * BK + kh * 16];
    const uint32_t lbBase = (uint32_t)(uintptr_t)&sB[0][tid * BK];
    const uint32_t aStride = (uint32_t)(BM * BK * sizeof(_Float16));    // 8 KB
    const uint32_t bStride = (uint32_t)(BN_G * BK * sizeof(_Float16)); // 16 KB

    v8f acc[4][4];
#pragma unroll
    for (int i = 0; i < 4; ++i)
#pragma unroll
        for (int j = 0; j < 4; ++j)
            acc[i][j] = {};

    // Prologue: fill 3 of the 4 buffers.
#pragma unroll
    for (int t = 0; t < 3; ++t)
        issue_tile(laBase + (uint32_t)t * aStride, aSrc + t * BK,
                   lbBase + (uint32_t)t * bStride, bSrc + t * BK);
    asm volatile("s_wait_asynccnt 12" ::: "memory"); // tile 0 resident
    __syncthreads();

    const int KT = K_DIM / BK;
    for (int kt = 0; kt < KT; ++kt) {
        const int buf = kt & (DEPTH - 1);

        // Issue tile kt+3 into the buffer drained at the kt-1 barrier.
        if (kt + 3 < KT) {
            const uint32_t ib = (uint32_t)((kt + 3) & (DEPTH - 1));
            issue_tile(laBase + ib * aStride, aSrc + (kt + 3) * BK,
                       lbBase + ib * bStride, bSrc + (kt + 3) * BK);
        }

        // Fragments: single contiguous 32B LDS read each (A, B: k = 16*half + s).
        v16h afr[4];
#pragma unroll
        for (int i = 0; i < 4; ++i)
            afr[i] = *(const v16h*)&sA[buf][(waveM * 64 + i * 16 + fr) * BK + half * 16];
        v16h bfr[4];
#pragma unroll
        for (int j = 0; j < 4; ++j)
            bfr[j] = *(const v16h*)&sB[buf][(waveN * 64 + j * 16 + fr) * BK + half * 16];

#pragma unroll
        for (int i = 0; i < 4; ++i)
#pragma unroll
            for (int j = 0; j < 4; ++j)
                acc[i][j] = __builtin_amdgcn_wmma_f32_16x16x32_f16(
                    false, afr[i], false, bfr[j], (short)0, acc[i][j], false, false);

        // tile kt+1 resident for every wave after this barrier.
        asm volatile("s_wait_asynccnt 12" ::: "memory");
        __syncthreads();
    }

    // Epilogue: C layout — VGPR r, lane l -> M = r + 8*(l>=16), N = l%16.
#pragma unroll
    for (int j = 0; j < 4; ++j) {
        const int col = bN + waveN * 64 + j * 16 + fr;
        const float bv = bias[col];
#pragma unroll
        for (int i = 0; i < 4; ++i) {
            const int rbase = bM + waveM * 64 + i * 16 + half * 8;
#pragma unroll
            for (int r = 0; r < 8; ++r)
                out[(size_t)(rbase + r) * N_DIM + col] = acc[i][j][r] + bv;
        }
    }
}

// ---------------------------------------------------------------------------
// Fallback: fused dequant + GEMM (128x128 tile), used if scratch too small
// ---------------------------------------------------------------------------
__global__ __launch_bounds__(256)
void nvfp4_linear_fused(const float* __restrict__ x,
                        const int*   __restrict__ wp,
                        const float* __restrict__ ws,
                        const float* __restrict__ bias,
                        float* __restrict__ out) {
    __shared__ __align__(32) _Float16 sA[BM * BK];
    __shared__ __align__(32) _Float16 sB[128 * BK];

    const int tid   = threadIdx.x;
    const int lane  = tid & 31;
    const int wave  = tid >> 5;
    const int waveM = wave & 3;
    const int waveN = wave >> 2;
    const int half  = lane >> 4;
    const int fr    = lane & 15;
    const int bM    = blockIdx.y * BM;
    const int bN    = blockIdx.x * 128;
    const int row   = tid >> 1;
    const int kh    = tid & 1;

    const float* aSrc = x  + (size_t)(bM + row) * K_DIM + kh * 16;
    const int*   wSrc = wp + (size_t)(bN + row) * (K_DIM / 2) + kh * 8;
    const float* sSrc = ws + (size_t)(bN + row) * (K_DIM / 16) + kh;

    v8f acc[2][4];
#pragma unroll
    for (int i = 0; i < 2; ++i)
#pragma unroll
        for (int j = 0; j < 4; ++j)
            acc[i][j] = {};

    for (int kt = 0; kt < K_DIM / BK; ++kt) {
        const float4* ap = (const float4*)(aSrc + kt * BK);
        float4 a0 = ap[0], a1 = ap[1], a2 = ap[2], a3 = ap[3];
        const int4* wq = (const int4*)(wSrc + kt * 16);
        int4 w0 = wq[0], w1 = wq[1];
        const float scale = sSrc[kt * 2];

        if (kt + 1 < K_DIM / BK)
            __builtin_prefetch(aSrc + (kt + 1) * BK, 0, 3);

        float af[16] = {a0.x, a0.y, a0.z, a0.w, a1.x, a1.y, a1.z, a1.w,
                        a2.x, a2.y, a2.z, a2.w, a3.x, a3.y, a3.z, a3.w};
        v8h aLo, aHi;
#pragma unroll
        for (int e = 0; e < 8; ++e) {
            aLo[e] = (_Float16)af[e];
            aHi[e] = (_Float16)af[e + 8];
        }

        const int wb[8] = {w0.x, w0.y, w0.z, w0.w, w1.x, w1.y, w1.z, w1.w};
        const _Float16 hs = (_Float16)scale;
        v8h wLo, wHi;
#pragma unroll
        for (int e = 0; e < 16; ++e) {
            const int byte = wb[e >> 1];
            const int code = (e & 1) ? (byte & 15) : ((byte >> 4) & 15);
            const uint16_t hb = (uint16_t)e2m1_f16_bits(code);
            _Float16 hm;
            __builtin_memcpy(&hm, &hb, sizeof(hm));
            const _Float16 hv = hm * hs;
            if (e < 8) wLo[e] = hv; else wHi[e - 8] = hv;
        }

        __syncthreads();
        *(v8h*)&sA[row * BK + kh * 16]     = aLo;
        *(v8h*)&sA[row * BK + kh * 16 + 8] = aHi;
        *(v8h*)&sB[row * BK + kh * 16]     = wLo;
        *(v8h*)&sB[row * BK + kh * 16 + 8] = wHi;
        __syncthreads();

        v16h afr[2];
#pragma unroll
        for (int i = 0; i < 2; ++i) {
            const _Float16* p = &sA[(waveM * 32 + i * 16 + fr) * BK + half * 16];
            v8h lo = *(const v8h*)p;
            v8h hi = *(const v8h*)(p + 8);
            afr[i] = __builtin_shufflevector(lo, hi, 0, 1, 2, 3, 4, 5, 6, 7,
                                                     8, 9, 10, 11, 12, 13, 14, 15);
        }
        v16h bfr[4];
#pragma unroll
        for (int j = 0; j < 4; ++j)
            bfr[j] = *(const v16h*)&sB[(waveN * 64 + j * 16 + fr) * BK + half * 16];

#pragma unroll
        for (int i = 0; i < 2; ++i)
#pragma unroll
            for (int j = 0; j < 4; ++j)
                acc[i][j] = __builtin_amdgcn_wmma_f32_16x16x32_f16(
                    false, afr[i], false, bfr[j], (short)0, acc[i][j], false, false);
    }

#pragma unroll
    for (int j = 0; j < 4; ++j) {
        const int col = bN + waveN * 64 + j * 16 + fr;
        const float bv = bias[col];
#pragma unroll
        for (int i = 0; i < 2; ++i) {
            const int rbase = bM + waveM * 32 + i * 16 + half * 8;
#pragma unroll
            for (int r = 0; r < 8; ++r)
                out[(size_t)(rbase + r) * N_DIM + col] = acc[i][j][r] + bv;
        }
    }
}

// ---------------------------------------------------------------------------
extern "C" void kernel_launch(void* const* d_in, const int* in_sizes, int n_in,
                              void* d_out, int out_size, void* d_ws, size_t ws_size,
                              hipStream_t stream) {
    const float* x    = (const float*)d_in[0];
    const int*   wp   = (const int*)d_in[1];
    const float* ws   = (const float*)d_in[2];
    const float* bias = (const float*)d_in[3];
    float*       out  = (float*)d_out;

    const int M = in_sizes[0] / K_DIM; // 8192
    const size_t xhElems = (size_t)M * K_DIM;
    const size_t whElems = (size_t)N_DIM * K_DIM;
    const size_t needed  = (xhElems + whElems) * sizeof(_Float16); // 96 MB

    if (ws_size >= needed) {
        _Float16* xh = (_Float16*)d_ws;
        _Float16* wh = xh + xhElems;
        cvt_x_f16<<<(int)(xhElems / (256 * 8)), 256, 0, stream>>>(x, xh);
        dequant_w_f16<<<(int)(whElems / 16 / 256), 256, 0, stream>>>(wp, ws, wh);
        dim3 gridG(N_DIM / BN_G, M / BM);
        gemm_f16_wmma<<<gridG, 256, 0, stream>>>(xh, wh, bias, out);
    } else {
        dim3 gridF(N_DIM / 128, M / BM);
        nvfp4_linear_fused<<<gridF, 256, 0, stream>>>(x, wp, ws, bias, out);
    }
}
